// AttentionIAM_84774064488487
// MI455X (gfx1250) — matrix operand order repaired and checked by reading, not compiled, not run on hardware
//
#include <hip/hip_runtime.h>
#include <hip/hip_bf16.h>

typedef __attribute__((ext_vector_type(16))) _Float16 v16h;
typedef __attribute__((ext_vector_type(8)))  _Float16 v8h;
typedef __attribute__((ext_vector_type(8)))  float    v8f;
typedef int v4i_vs __attribute__((vector_size(16)));   // matches builtin param type

constexpr int   kB   = 8;
constexpr int   kC   = 512;
constexpr int   kT   = 1024;   // H*W = 32*32
constexpr int   kNH  = 8;
constexpr int   kCH  = 64;     // head dim
constexpr float kScale = 0.35355339059327373f;  // 64^-0.25
constexpr float kEps   = 1e-5f;

// Padded row pitch for the LDS weight strip: 512 + 8 halves = 1040 B/row.
// 1040/4 = 260 dwords -> consecutive rows land 4 banks apart (64-bank LDS),
// so the 16 A-rows read by a wave hit 16 distinct banks.
constexpr int kAPitch = 520;

#if defined(__gfx1250__) && \
    __has_builtin(__builtin_amdgcn_global_load_async_to_lds_b128) && \
    __has_builtin(__builtin_amdgcn_s_wait_asynccnt)
#define HAVE_ASYNC_LDS 1
#else
#define HAVE_ASYNC_LDS 0
#endif

__device__ __forceinline__ v8f wmma_f16(v16h a, v16h b, v8f c) {
  // D = A(16x32 f16) x B(32x16 f16) + C(16x16 f32)
  return __builtin_amdgcn_wmma_f32_16x16x32_f16(false, a, false, b, (short)0, c,
                                                false, false);
}

// Stage a 16-row x 512-col f16 strip (row-major, 1024B rows) into LDS with
// 16B row padding. 256 threads: 16 threads/row, 64B each, 4x16B transfers.
__device__ __forceinline__ void stage_A16x512(_Float16* lds, const _Float16* gsrc) {
  int tid = threadIdx.x;
  int row = tid >> 4, seg = tid & 15;
#if HAVE_ASYNC_LDS
  char* s = (char*)(gsrc + (size_t)row * kC) + seg * 64;
  char* d = (char*)(lds + (size_t)row * kAPitch) + seg * 64;
#pragma unroll
  for (int i = 0; i < 4; ++i) {
    __builtin_amdgcn_global_load_async_to_lds_b128(
        (v4i_vs*)(s + i * 16), (v4i_vs*)(d + i * 16), 0, 0);
  }
  __builtin_amdgcn_s_wait_asynccnt(0);
  __syncthreads();
#else
  const float4* s = (const float4*)((const char*)(gsrc + (size_t)row * kC) + seg * 64);
  float4* d = (float4*)((char*)(lds + (size_t)row * kAPitch) + seg * 64);
#pragma unroll
  for (int i = 0; i < 4; ++i) d[i] = s[i];
  __syncthreads();
#endif
}

// Build the 16x32 A-fragment for this wave from the LDS strip.
__device__ __forceinline__ v16h a_frag_lds(const _Float16* lds, int ln, int hf, int k0) {
  union { v16h v; v8h h[2]; } A;
  const _Float16* ar = lds + (size_t)ln * kAPitch + k0;
  A.h[0] = *(const v8h*)(ar + hf * 8);
  A.h[1] = *(const v8h*)(ar + 16 + hf * 8);
  return A.v;
}

// ---------------------------------------------------------------------------
// Kernel 1: convert weights to f16; fold q/k scale (64^-0.25) into qkv rows.
// ---------------------------------------------------------------------------
__global__ void prep_w_k(const float* __restrict__ qw, const float* __restrict__ pw,
                         _Float16* __restrict__ qw16, _Float16* __restrict__ pw16) {
  int i = blockIdx.x * 256 + threadIdx.x;
  constexpr int NQ = 1536 * 512;
  constexpr int NP = 512 * 512;
  if (i < NQ) {
    int o = i >> 9;  // row (output channel), 512 cols per row
    float s = (o < 1024) ? kScale : 1.0f;
    qw16[i] = (_Float16)(qw[i] * s);
  } else if (i - NQ < NP) {
    int j = i - NQ;
    pw16[j] = (_Float16)pw[j];
  }
}

// ---------------------------------------------------------------------------
// Kernel 2: GroupNorm statistics. One block per (b, group): 16384 contiguous
// floats -> mean, rstd.
// ---------------------------------------------------------------------------
__global__ void gn_stats_k(const float* __restrict__ x, float* __restrict__ stats) {
  __shared__ float s1[256], s2[256];
  int grp = blockIdx.x;  // b*32 + g, contiguous 16*1024-element region
  const float4* p = (const float4*)(x + (size_t)grp * 16384);
  float a = 0.f, b2 = 0.f;
  for (int i = threadIdx.x; i < 4096; i += 256) {
    float4 v4 = p[i];
    a  += v4.x + v4.y + v4.z + v4.w;
    b2 += v4.x * v4.x + v4.y * v4.y + v4.z * v4.z + v4.w * v4.w;
  }
  s1[threadIdx.x] = a; s2[threadIdx.x] = b2;
  __syncthreads();
  for (int st = 128; st; st >>= 1) {
    if (threadIdx.x < st) {
      s1[threadIdx.x] += s1[threadIdx.x + st];
      s2[threadIdx.x] += s2[threadIdx.x + st];
    }
    __syncthreads();
  }
  if (threadIdx.x == 0) {
    float mean = s1[0] * (1.0f / 16384.0f);
    float var  = s2[0] * (1.0f / 16384.0f) - mean * mean;
    stats[grp * 2 + 0] = mean;
    stats[grp * 2 + 1] = rsqrtf(var + kEps);
  }
}

// ---------------------------------------------------------------------------
// Kernel 3: apply GN + affine, transpose [b,c,t] f32 -> [b,t,c] f16 via LDS.
// ---------------------------------------------------------------------------
__global__ void gn_apply_k(const float* __restrict__ x, const float* __restrict__ stats,
                           const float* __restrict__ gw, const float* __restrict__ gb,
                           _Float16* __restrict__ nT) {
  __shared__ float tile[32][33];
  int b = blockIdx.z, c0 = blockIdx.x * 32, t0 = blockIdx.y * 32;
  int tx = threadIdx.x, ty = threadIdx.y;
#pragma unroll
  for (int r = 0; r < 4; ++r) {
    int cl = ty + r * 8;
    int c = c0 + cl;
    int g = c >> 4;  // 16 channels per group
    float mean = stats[(b * 32 + g) * 2 + 0];
    float rstd = stats[(b * 32 + g) * 2 + 1];
    float v = x[((size_t)(b * kC + c)) * kT + t0 + tx];
    tile[cl][tx] = (v - mean) * rstd * gw[c] + gb[c];
  }
  __syncthreads();
#pragma unroll
  for (int r = 0; r < 4; ++r) {
    int tl = ty + r * 8;
    nT[((size_t)(b * kT) + t0 + tl) * kC + c0 + tx] = (_Float16)tile[tx][tl];
  }
}

// ---------------------------------------------------------------------------
// Kernel 4: QKV GEMM. qkv[o,t] = sum_c W16[o,c] * nT[t,c] (+ bias).
// Block: 16 output rows, 256 t-cols. A strip staged (async) into LDS once and
// shared by all 8 waves; each wave = two 16x16 tiles sharing one A fragment.
// q,k -> [bh, t, ch] f16 (scale pre-folded); v -> [bh, ch, t] f16.
// ---------------------------------------------------------------------------
__global__ void __launch_bounds__(256)
qkv_gemm_k(const _Float16* __restrict__ W, const _Float16* __restrict__ X,
           const float* __restrict__ bias, _Float16* __restrict__ q16,
           _Float16* __restrict__ k16, _Float16* __restrict__ v16) {
  __shared__ _Float16 Atile[16 * kAPitch];  // ~16.6 KB
  int o0 = blockIdx.x * 16;
  int b  = blockIdx.z;
  int w  = threadIdx.x >> 5, lane = threadIdx.x & 31;
  int ln = lane & 15, hf = lane >> 4;
  int t0 = blockIdx.y * 256 + w * 32;

  stage_A16x512(Atile, W + (size_t)o0 * kC);

  const _Float16* Xr0 = X + ((size_t)b * kT + t0 + ln) * kC;       // B col N=ln
  const _Float16* Xr1 = Xr0 + (size_t)16 * kC;                     // t-tile +16
  v8f acc0 = {0.f, 0.f, 0.f, 0.f, 0.f, 0.f, 0.f, 0.f};
  v8f acc1 = acc0;
  for (int k0 = 0; k0 < kC; k0 += 32) {
    v16h Af  = a_frag_lds(Atile, ln, hf, k0);
    union { v16h v; v8h h[2]; } B0, B1;
    B0.h[0] = *(const v8h*)(Xr0 + k0 + hf * 8);
    B0.h[1] = *(const v8h*)(Xr0 + k0 + 16 + hf * 8);
    B1.h[0] = *(const v8h*)(Xr1 + k0 + hf * 8);
    B1.h[1] = *(const v8h*)(Xr1 + k0 + 16 + hf * 8);
    v16h Bf0 = B0.v;
    v16h Bf1 = B1.v;
    __builtin_prefetch(Xr0 + k0 + 64, 0, 3);
    __builtin_prefetch(Xr1 + k0 + 64, 0, 3);
    acc0 = wmma_f16(Af, Bf0, acc0);
    acc1 = wmma_f16(Af, Bf1, acc1);
  }
  v8f accs[2] = {acc0, acc1};
#pragma unroll
  for (int tt = 0; tt < 2; ++tt) {
    int t = t0 + tt * 16 + ln;
    if (o0 < 1024) {               // q or k -> [bh, t, ch], packed 16B store
      _Float16* dst = (o0 < 512) ? q16 : k16;
      int oo  = o0 & 511;
      int bh  = b * kNH + (oo >> 6);
      int chb = (oo & 63) + hf * 8;
      v8h outv;
#pragma unroll
      for (int j = 0; j < 8; ++j) {
        int o = o0 + j + hf * 8;
        outv[j] = (_Float16)(accs[tt][j] + bias[o] * kScale);
      }
      *(v8h*)(dst + ((size_t)bh * kT + t) * kCH + chb) = outv;
    } else {                       // v -> [bh, ch, t]
      int ov = o0 - 1024;
      int bh = b * kNH + (ov >> 6);
#pragma unroll
      for (int j = 0; j < 8; ++j) {
        int o  = o0 + j + hf * 8;
        int ch = (ov & 63) + j + hf * 8;
        v16[((size_t)bh * kCH + ch) * kT + t] = (_Float16)(accs[tt][j] + bias[o]);
      }
    }
  }
}

// ---------------------------------------------------------------------------
// Kernel 5: attention. One block per (bh, 16 query rows). 64KB f32 score row
// strip in dynamic LDS; full softmax; P@V^T with cross-wave LDS reduction.
// Output a -> aBuf [b, c, t] f32 (contiguous per-lane stores).
// ---------------------------------------------------------------------------
__global__ void __launch_bounds__(256)
attn_k(const _Float16* __restrict__ Q, const _Float16* __restrict__ K,
       const _Float16* __restrict__ V, float* __restrict__ aOut) {
  extern __shared__ float smem[];
  float* sc   = smem;              // 16 * 1024 scores/probs
  float* pbuf = smem + 16 * kT;    // 4 * 256 partial tiles

  int bh = blockIdx.x;
  int t0 = blockIdx.y * 16;
  int w = threadIdx.x >> 5, lane = threadIdx.x & 31;
  int ln = lane & 15, hf = lane >> 4;

  const _Float16* Qb = Q + ((size_t)bh * kT) * kCH;
  const _Float16* Kb = K + ((size_t)bh * kT) * kCH;

  // ---- Phase 1: scores[t, s] = q[t,:] . k[s,:] (scale pre-folded) ----
  // Q fragments are invariant across s-tiles: load once.
  v16h Aq[2];
#pragma unroll
  for (int kk = 0; kk < 2; ++kk) {
    union { v16h v; v8h h[2]; } A;
    const _Float16* qp = Qb + (size_t)(t0 + ln) * kCH + kk * 32;
    A.h[0] = *(const v8h*)(qp + hf * 8);
    A.h[1] = *(const v8h*)(qp + 16 + hf * 8);
    Aq[kk] = A.v;
  }
  for (int st = w; st < 64; st += 8) {
    int s0 = st * 16;
    v8f acc = {0.f, 0.f, 0.f, 0.f, 0.f, 0.f, 0.f, 0.f};
#pragma unroll
    for (int kk = 0; kk < 2; ++kk) {
      union { v16h v; v8h h[2]; } B;
      const _Float16* kp = Kb + (size_t)(s0 + ln) * kCH + kk * 32;
      B.h[0] = *(const v8h*)(kp + hf * 8);
      B.h[1] = *(const v8h*)(kp + 16 + hf * 8);
      v16h Bf = B.v;
      acc = wmma_f16(Aq[kk], Bf, acc);
    }
#pragma unroll
    for (int j = 0; j < 8; ++j)
      sc[(j + hf * 8) * kT + s0 + ln] = acc[j];
  }
  __syncthreads();

  // ---- Phase 2: row softmax (f32, wave32 shuffle reductions) ----
#pragma unroll
  for (int rr = 0; rr < 2; ++rr) {
    int r = w * 2 + rr;
    float* row = sc + r * kT;
    float mx = -3.4e38f;
    for (int s = lane; s < kT; s += 32) mx = fmaxf(mx, row[s]);
#pragma unroll
    for (int off = 16; off; off >>= 1) mx = fmaxf(mx, __shfl_xor(mx, off, 32));
    float sum = 0.f;
    for (int s = lane; s < kT; s += 32) {
      float e = __expf(row[s] - mx);
      row[s] = e;
      sum += e;
    }
#pragma unroll
    for (int off = 16; off; off >>= 1) sum += __shfl_xor(sum, off, 32);
    float inv = 1.f / sum;
    for (int s = lane; s < kT; s += 32) row[s] *= inv;
  }
  __syncthreads();

  // ---- Phase 3: a[t, ch] = sum_s P[t,s] * v[ch,s] ----
  int ct = w & 3;   // ch-tile 0..3
  int kh = w >> 2;  // K half
  const _Float16* Vb = V + ((size_t)bh * kCH) * kT;
  const float* prow = sc + ln * kT;  // A row M=ln
  v8f acc = {0.f, 0.f, 0.f, 0.f, 0.f, 0.f, 0.f, 0.f};
  for (int s0 = kh * 512; s0 < kh * 512 + 512; s0 += 32) {
    v16h Af;
    const float* p0 = prow + s0 + hf * 8;
    const float* p1 = prow + s0 + 16 + hf * 8;
#pragma unroll
    for (int i = 0; i < 8; ++i) {
      Af[i]     = (_Float16)p0[i];
      Af[8 + i] = (_Float16)p1[i];
    }
    union { v16h v; v8h h[2]; } B;
    const _Float16* vp = Vb + (size_t)(ct * 16 + ln) * kT + s0;
    B.h[0] = *(const v8h*)(vp + hf * 8);
    B.h[1] = *(const v8h*)(vp + 16 + hf * 8);
    v16h Bf = B.v;
    acc = wmma_f16(Af, Bf, acc);
  }
  if (w >= 4) {
#pragma unroll
    for (int j = 0; j < 8; ++j) pbuf[(w - 4) * 256 + j * 32 + lane] = acc[j];
  }
  __syncthreads();
  if (w < 4) {
    int b = bh >> 3, hd = bh & 7;
    int c = hd * kCH + ct * 16 + ln;        // N = ln -> channel
    float* dst = aOut + ((size_t)(b * kC + c)) * kT + t0 + hf * 8;
#pragma unroll
    for (int j = 0; j < 8; ++j)
      dst[j] = acc[j] + pbuf[w * 256 + j * 32 + lane];
  }
}

// ---------------------------------------------------------------------------
// Kernel 6: residual (x + a) and transpose to [b, t, c] f16 via LDS.
// ---------------------------------------------------------------------------
__global__ void resid_k(const float* __restrict__ x, const float* __restrict__ aBuf,
                        _Float16* __restrict__ r16) {
  __shared__ float tile[32][33];
  int b = blockIdx.z, c0 = blockIdx.x * 32, t0 = blockIdx.y * 32;
  int tx = threadIdx.x, ty = threadIdx.y;
#pragma unroll
  for (int r = 0; r < 4; ++r) {
    int cl = ty + r * 8;
    size_t idx = ((size_t)(b * kC + c0 + cl)) * kT + t0 + tx;
    tile[cl][tx] = x[idx] + aBuf[idx];
  }
  __syncthreads();
#pragma unroll
  for (int r = 0; r < 4; ++r) {
    int tl = ty + r * 8;
    r16[((size_t)(b * kT) + t0 + tl) * kC + c0 + tx] = (_Float16)tile[tx][tl];
  }
}

// ---------------------------------------------------------------------------
// Kernel 7: proj GEMM. out[o,t] = sum_c P16[o,c] * r16[t,c] + proj_b[o], f32.
// Same structure as the QKV GEMM (LDS-staged A, dual accumulators).
// ---------------------------------------------------------------------------
__global__ void __launch_bounds__(256)
proj_gemm_k(const _Float16* __restrict__ W, const _Float16* __restrict__ X,
            const float* __restrict__ bias, float* __restrict__ out) {
  __shared__ _Float16 Atile[16 * kAPitch];
  int o0 = blockIdx.x * 16;
  int b  = blockIdx.z;
  int w  = threadIdx.x >> 5, lane = threadIdx.x & 31;
  int ln = lane & 15, hf = lane >> 4;
  int t0 = blockIdx.y * 256 + w * 32;

  stage_A16x512(Atile, W + (size_t)o0 * kC);

  const _Float16* Xr0 = X + ((size_t)b * kT + t0 + ln) * kC;
  const _Float16* Xr1 = Xr0 + (size_t)16 * kC;
  v8f acc0 = {0.f, 0.f, 0.f, 0.f, 0.f, 0.f, 0.f, 0.f};
  v8f acc1 = acc0;
  for (int k0 = 0; k0 < kC; k0 += 32) {
    v16h Af  = a_frag_lds(Atile, ln, hf, k0);
    union { v16h v; v8h h[2]; } B0, B1;
    B0.h[0] = *(const v8h*)(Xr0 + k0 + hf * 8);
    B0.h[1] = *(const v8h*)(Xr0 + k0 + 16 + hf * 8);
    B1.h[0] = *(const v8h*)(Xr1 + k0 + hf * 8);
    B1.h[1] = *(const v8h*)(Xr1 + k0 + 16 + hf * 8);
    v16h Bf0 = B0.v;
    v16h Bf1 = B1.v;
    __builtin_prefetch(Xr0 + k0 + 64, 0, 3);
    __builtin_prefetch(Xr1 + k0 + 64, 0, 3);
    acc0 = wmma_f16(Af, Bf0, acc0);
    acc1 = wmma_f16(Af, Bf1, acc1);
  }
  v8f accs[2] = {acc0, acc1};
#pragma unroll
  for (int tt = 0; tt < 2; ++tt) {
    int t = t0 + tt * 16 + ln;
#pragma unroll
    for (int j = 0; j < 8; ++j) {
      int o = o0 + j + hf * 8;
      out[((size_t)(b * kC + o)) * kT + t] = accs[tt][j] + bias[o];
    }
  }
}

// ---------------------------------------------------------------------------
extern "C" void kernel_launch(void* const* d_in, const int* in_sizes, int n_in,
                              void* d_out, int out_size, void* d_ws, size_t ws_size,
                              hipStream_t stream) {
  (void)in_sizes; (void)n_in; (void)out_size; (void)ws_size;
  const float* x      = (const float*)d_in[0];
  const float* gn_w   = (const float*)d_in[1];
  const float* gn_b   = (const float*)d_in[2];
  const float* qkv_w  = (const float*)d_in[3];
  const float* qkv_b  = (const float*)d_in[4];
  const float* proj_w = (const float*)d_in[5];
  const float* proj_b = (const float*)d_in[6];
  float* out = (float*)d_out;

  char* ws = (char*)d_ws;
  size_t off = 0;
  auto take = [&](size_t bytes) -> char* {
    char* p = ws + off;
    off = (off + bytes + 255) & ~(size_t)255;
    return p;
  };
  float*    stats = (float*)take(256 * 2 * sizeof(float));
  _Float16* qw16  = (_Float16*)take((size_t)1536 * 512 * 2);
  _Float16* pw16  = (_Float16*)take((size_t)512 * 512 * 2);
  _Float16* nT    = (_Float16*)take((size_t)kB * kT * kC * 2);
  _Float16* q16   = (_Float16*)take((size_t)kB * kNH * kT * kCH * 2);
  _Float16* k16   = (_Float16*)take((size_t)kB * kNH * kT * kCH * 2);
  _Float16* v16   = (_Float16*)take((size_t)kB * kNH * kCH * kT * 2);
  float*    aBuf  = (float*)take((size_t)kB * kC * kT * sizeof(float));
  _Float16* r16   = (_Float16*)take((size_t)kB * kT * kC * 2);

  prep_w_k<<<4096, 256, 0, stream>>>(qkv_w, proj_w, qw16, pw16);
  gn_stats_k<<<256, 256, 0, stream>>>(x, stats);
  gn_apply_k<<<dim3(16, 32, 8), dim3(32, 8), 0, stream>>>(x, stats, gn_w, gn_b, nT);
  qkv_gemm_k<<<dim3(96, 4, 8), 256, 0, stream>>>(qw16, nT, qkv_b, q16, k16, v16);
  size_t attn_lds = (size_t)(16 * kT + 4 * 256) * sizeof(float);  // 68 KB
  attn_k<<<dim3(64, 64), 256, attn_lds, stream>>>(q16, k16, v16, aBuf);
  resid_k<<<dim3(16, 32, 8), dim3(32, 8), 0, stream>>>(x, aBuf, r16);
  proj_gemm_k<<<dim3(32, 4, 8), 256, 0, stream>>>(pw16, r16, proj_b, out);
}
